// MambaWorldModel_17403207483679
// MI455X (gfx1250) — matrix operand order repaired and checked by reading, not compiled, not run on hardware
//
#include <hip/hip_runtime.h>
#include <hip/hip_bf16.h>
#include <cstdint>
#include <cstddef>

// ---- problem constants ----
#define BATCH   8192
#define OBS     512
#define ACT     256
#define HID     1024
#define D_INNER 2048
#define D_STATE 64
#define NHEADS  32
#define CONV_DIM (D_INNER + 2 * D_STATE)             // 2176
#define D_PROJ  (2 * D_INNER + 2 * D_STATE + NHEADS) // 4256
#define RMS_EPS 1e-5f

typedef unsigned short u16t;
typedef __attribute__((ext_vector_type(16))) __bf16          v16bf;
typedef __attribute__((ext_vector_type(2)))  __bf16          v2bf;
typedef __attribute__((ext_vector_type(8)))  float           v8f;
typedef __attribute__((ext_vector_type(8)))  unsigned short  us8;
typedef __attribute__((ext_vector_type(16))) unsigned short  us16;
typedef __attribute__((ext_vector_type(4)))  int             v4i;

// Async global->LDS path (CDNA5): guarded so the file compiles on either toolchain.
#if defined(__has_builtin)
#  if __has_builtin(__builtin_amdgcn_global_load_async_to_lds_b128)
#    define USE_ASYNC_LDS 1
#  endif
#  if __has_builtin(__builtin_amdgcn_cvt_pk_bf16_f32)
#    define USE_CVT_PK_BF16 1
#  endif
#endif
#ifndef USE_ASYNC_LDS
#  define USE_ASYNC_LDS 0
#endif
#ifndef USE_CVT_PK_BF16
#  define USE_CVT_PK_BF16 0
#endif

#if USE_ASYNC_LDS
// Builtin signature (confirmed via clang diagnostics): param0 = AS1 int4*,
// param1 = AS3 int4*, then two immediate ints (offset, cpol).
typedef __attribute__((address_space(1))) v4i* gbl_v4ip;
typedef __attribute__((address_space(3))) v4i* lds_v4ip;
__device__ __forceinline__ void wait_async0() {
#  if __has_builtin(__builtin_amdgcn_s_wait_asynccnt)
  __builtin_amdgcn_s_wait_asynccnt(0);
#  else
  asm volatile("s_wait_asynccnt 0x0" ::: "memory");
#  endif
}
#endif

// ---- helpers ----
__device__ __forceinline__ unsigned short f2bf(float f) {
  unsigned u = __builtin_bit_cast(unsigned, f);
  u = u + 0x7FFFu + ((u >> 16) & 1u);            // round-to-nearest-even
  return (unsigned short)(u >> 16);
}
__device__ __forceinline__ unsigned pack_bf2(float a, float b) {
#if USE_CVT_PK_BF16
  v2bf r = __builtin_amdgcn_cvt_pk_bf16_f32(a, b);   // single v_cvt_pk_bf16_f32
  return __builtin_bit_cast(unsigned, r);
#else
  unsigned ua = __builtin_bit_cast(unsigned, a);
  unsigned ub = __builtin_bit_cast(unsigned, b);
  ua = ua + 0x7FFFu + ((ua >> 16) & 1u);
  ub = ub + 0x7FFFu + ((ub >> 16) & 1u);
  return (ua >> 16) | (ub & 0xFFFF0000u);
#endif
}
__device__ __forceinline__ float silu_f(float x) { return x / (1.0f + __expf(-x)); }

// A-fragment: elements 0..7 at p[0..7], elements 8..15 at p[16..23] (K-block gap 16)
__device__ __forceinline__ v16bf frag_gap16(const u16t* p) {
  us8 lo = *(const us8*)p;
  us8 hi = *(const us8*)(p + 16);
  us16 c;
#pragma unroll
  for (int i = 0; i < 8; ++i) { c[i] = lo[i]; c[i + 8] = hi[i]; }
  return __builtin_bit_cast(v16bf, c);
}
// B-fragment: 16 contiguous K elements per lane
__device__ __forceinline__ v16bf frag_cont16(const u16t* p) {
  us8 lo = *(const us8*)p;
  us8 hi = *(const us8*)(p + 8);
  us16 c;
#pragma unroll
  for (int i = 0; i < 8; ++i) { c[i] = lo[i]; c[i + 8] = hi[i]; }
  return __builtin_bit_cast(v16bf, c);
}

// ---- concat [obs | action] -> f32 Xcat[B,768] ----
__global__ void concat_kernel(const float* __restrict__ obs,
                              const float* __restrict__ act,
                              float* __restrict__ out) {
  int i = blockIdx.x * blockDim.x + threadIdx.x;
  int total = BATCH * (OBS + ACT);
  if (i >= total) return;
  int b = i / (OBS + ACT);
  int c = i - b * (OBS + ACT);
  out[i] = (c < OBS) ? obs[(size_t)b * OBS + c] : act[(size_t)b * ACT + (c - OBS)];
}

// ---- transpose + f32->bf16: W[K,N] -> Wt[N,K] (bf16) ----
__global__ void transpose_cvt_kernel(const float* __restrict__ W,
                                     u16t* __restrict__ Wt, int K, int N) {
  __shared__ float tile[32][33];
  int tx = threadIdx.x, ty = threadIdx.y;
  int n0 = blockIdx.x * 32, k0 = blockIdx.y * 32;
#pragma unroll
  for (int u = 0; u < 4; ++u) {
    int k = k0 + ty + 8 * u, n = n0 + tx;
    tile[ty + 8 * u][tx] = (k < K && n < N) ? W[(size_t)k * N + n] : 0.0f;
  }
  __syncthreads();
#pragma unroll
  for (int u = 0; u < 4; ++u) {
    int n = n0 + ty + 8 * u, k = k0 + tx;
    if (n < N && k < K) Wt[(size_t)n * K + k] = f2bf(tile[tx][ty + 8 * u]);
  }
}

// ---- generic bf16 WMMA GEMM: C[M,N] = A[M,K](f32) * Bt[N,K](bf16) (+bias) ----
#define BM 128
#define BN 128
#define BK 32
#define LDT 40   // padded LDS row stride (elements); 80 bytes => 16B-aligned b128 frag loads

__global__ __launch_bounds__(256) void gemm_bf16_kernel(
    const float* __restrict__ A, const u16t* __restrict__ Bt,
    const float* __restrict__ bias, float* __restrict__ C,
    int M, int N, int K) {
  __shared__ u16t As[2][BM * LDT];
  __shared__ u16t Bs[2][BN * LDT];

  const int t    = threadIdx.x;
  const int lane = t & 31;
  const int wave = t >> 5;
  const int wm   = wave >> 2;     // 0..1  (64 rows each)
  const int wn   = wave & 3;      // 0..3  (32 cols each)
  const int lo16 = lane & 15;
  const int hi   = lane >> 4;

  const int mBase = blockIdx.y * BM;
  const int nBase = blockIdx.x * BN;

  v8f acc[4][2] = {};

  const int ar   = t >> 3;          // A row in tile: 0..31 (+32*s)
  const int ac   = (t & 7) << 2;    // A col group (4 floats)
  const int bn   = t >> 2;          // B row (n) in tile: 0..63 (+64*s)
  const int bck  = (t & 3) << 3;    // B k offset (8 bf16 = 16B)

  // ---------- prologue: fill buffer 0 ----------
  float4 av0[4];
#pragma unroll
  for (int s = 0; s < 4; ++s)
    av0[s] = *(const float4*)&A[(size_t)(mBase + ar + 32 * s) * K + 0 + ac];
#if USE_ASYNC_LDS
#pragma unroll
  for (int s = 0; s < 2; ++s) {
    int ng = nBase + bn + 64 * s;
    if (ng < N)
      __builtin_amdgcn_global_load_async_to_lds_b128(
          (gbl_v4ip)&Bt[(size_t)ng * K + 0 + bck],
          (lds_v4ip)&Bs[0][(bn + 64 * s) * LDT + bck], 0, 0);
  }
#else
#pragma unroll
  for (int s = 0; s < 2; ++s) {
    int ng = nBase + bn + 64 * s;
    uint4 bv = (ng < N) ? *(const uint4*)&Bt[(size_t)ng * K + 0 + bck]
                        : make_uint4(0u, 0u, 0u, 0u);
    *(uint4*)&Bs[0][(bn + 64 * s) * LDT + bck] = bv;
  }
#endif
#pragma unroll
  for (int s = 0; s < 4; ++s) {
    uint2 p;
    p.x = pack_bf2(av0[s].x, av0[s].y);
    p.y = pack_bf2(av0[s].z, av0[s].w);
    *(uint2*)&As[0][(ar + 32 * s) * LDT + ac] = p;
  }
#if USE_ASYNC_LDS
  wait_async0();
#endif
  __syncthreads();

  // ---------- main loop: one barrier per K-step, loads overlap WMMA ----------
  int buf = 0;
  for (int kb = 0; kb < K; kb += BK) {
    const bool hasNext = (kb + BK) < K;
    float4 avn[4];
#if !USE_ASYNC_LDS
    uint4 bvn[2];
#endif
    if (hasNext) {
      const int kn = kb + BK;
#pragma unroll
      for (int s = 0; s < 4; ++s)
        avn[s] = *(const float4*)&A[(size_t)(mBase + ar + 32 * s) * K + kn + ac];
#if USE_ASYNC_LDS
#pragma unroll
      for (int s = 0; s < 2; ++s) {
        int ng = nBase + bn + 64 * s;
        if (ng < N)
          __builtin_amdgcn_global_load_async_to_lds_b128(
              (gbl_v4ip)&Bt[(size_t)ng * K + kn + bck],
              (lds_v4ip)&Bs[buf ^ 1][(bn + 64 * s) * LDT + bck], 0, 0);
      }
#else
#pragma unroll
      for (int s = 0; s < 2; ++s) {
        int ng = nBase + bn + 64 * s;
        bvn[s] = (ng < N) ? *(const uint4*)&Bt[(size_t)ng * K + kn + bck]
                          : make_uint4(0u, 0u, 0u, 0u);
      }
#endif
    }

    // compute current buffer
    {
      const u16t* Ab = As[buf];
      const u16t* Bb = Bs[buf];
      v16bf af[4], bfr[2];
#pragma unroll
      for (int i = 0; i < 4; ++i) {
        int m = wm * 64 + i * 16 + lo16;
        af[i] = frag_gap16(&Ab[m * LDT + hi * 8]);
      }
#pragma unroll
      for (int j = 0; j < 2; ++j) {
        int n = wn * 32 + j * 16 + lo16;
        bfr[j] = frag_gap16(&Bb[n * LDT + hi * 8]);
      }
#pragma unroll
      for (int i = 0; i < 4; ++i)
#pragma unroll
        for (int j = 0; j < 2; ++j)
          acc[i][j] = __builtin_amdgcn_wmma_f32_16x16x32_bf16(
              false, af[i], false, bfr[j], (short)0, acc[i][j], false, false);
    }

    if (hasNext) {
#pragma unroll
      for (int s = 0; s < 4; ++s) {
        uint2 p;
        p.x = pack_bf2(avn[s].x, avn[s].y);
        p.y = pack_bf2(avn[s].z, avn[s].w);
        *(uint2*)&As[buf ^ 1][(ar + 32 * s) * LDT + ac] = p;
      }
#if !USE_ASYNC_LDS
#pragma unroll
      for (int s = 0; s < 2; ++s)
        *(uint4*)&Bs[buf ^ 1][(bn + 64 * s) * LDT + bck] = bvn[s];
#endif
#if USE_ASYNC_LDS
      wait_async0();        // own async B loads for next buffer complete
#endif
      __syncthreads();      // everyone done reading cur + next buffer visible
      buf ^= 1;
    }
  }

  // epilogue (C/D layout: VGPR r -> M = (lane>>4)*8 + r, N = lane&15)
#pragma unroll
  for (int i = 0; i < 4; ++i) {
#pragma unroll
    for (int j = 0; j < 2; ++j) {
      int col = nBase + wn * 32 + j * 16 + lo16;
      if (col < N) {
        float bvv = bias ? bias[col] : 0.0f;
        int mrow = mBase + wm * 64 + i * 16 + hi * 8;
#pragma unroll
        for (int r = 0; r < 8; ++r)
          C[(size_t)(mrow + r) * N + col] = acc[i][j][r] + bvv;
      }
    }
  }
}

// ---- fused Mamba elementwise: zxbcdt[B,4256] -> y[B,2048] ----
__global__ __launch_bounds__(256) void mamba_ew_kernel(
    const float* __restrict__ Z, const float* __restrict__ conv_w,
    const float* __restrict__ conv_b, const float* __restrict__ dt_bias,
    const float* __restrict__ Dp, const float* __restrict__ rms_w,
    float* __restrict__ Y) {
  __shared__ float sh[D_INNER];
  __shared__ float red[256];
  __shared__ float coef[NHEADS];

  const int t = threadIdx.x;
  const size_t b = blockIdx.x;
  const float* zr = Z + b * D_PROJ;

  // bc = sum(silu(B)*silu(C)) over D_STATE
  float p = 0.0f;
  if (t < D_STATE) {
    int cB = D_INNER + t;            // conv index for B
    int cC = D_INNER + D_STATE + t;  // conv index for C
    float xb = zr[D_INNER + cB] * conv_w[cB * 4 + 3] + conv_b[cB];
    float xc = zr[D_INNER + cC] * conv_w[cC * 4 + 3] + conv_b[cC];
    p = silu_f(xb) * silu_f(xc);
  }
  red[t] = p;
  __syncthreads();
  for (int s = 128; s > 0; s >>= 1) {
    if (t < s) red[t] += red[t + s];
    __syncthreads();
  }
  float bc = red[0];
  if (t < NHEADS) {
    float d = zr[D_PROJ - NHEADS + t] + dt_bias[t];
    float sp = (d > 20.0f) ? d : log1pf(__expf(d));
    coef[t] = sp * bc + Dp[t];
  }
  __syncthreads();

  // y = coef[h] * silu(conv(x)) * silu(z); accumulate sum of squares
  float ss = 0.0f;
#pragma unroll
  for (int u = 0; u < 8; ++u) {
    int idx = t + 256 * u;
    float xa = zr[D_INNER + idx] * conv_w[idx * 4 + 3] + conv_b[idx];
    float yv = coef[idx >> 6] * silu_f(xa) * silu_f(zr[idx]);
    sh[idx] = yv;
    ss += yv * yv;
  }
  red[t] = ss;
  __syncthreads();
  for (int s = 128; s > 0; s >>= 1) {
    if (t < s) red[t] += red[t + s];
    __syncthreads();
  }
  float rstd = rsqrtf(red[0] / (float)D_INNER + RMS_EPS);
#pragma unroll
  for (int u = 0; u < 8; ++u) {
    int idx = t + 256 * u;
    Y[b * D_INNER + idx] = sh[idx] * rstd * rms_w[idx];
  }
}

// ---- reward: one wave per row, dot(h[b], W_rew)+b_rew ----
__global__ __launch_bounds__(256) void reward_kernel(
    const float* __restrict__ H, const float* __restrict__ wr,
    const float* __restrict__ br, float* __restrict__ out) {
  const int t = threadIdx.x;
  const int lane = t & 31, wave = t >> 5;
  const size_t b = (size_t)blockIdx.x * 8 + wave;
  float s = 0.0f;
#pragma unroll
  for (int u = 0; u < 32; ++u)
    s += H[b * HID + lane + 32 * u] * wr[lane + 32 * u];
#pragma unroll
  for (int off = 16; off > 0; off >>= 1)
    s += __shfl_xor(s, off, 32);
  if (lane == 0) out[b] = s + br[0];
}

// ---- launcher ----
extern "C" void kernel_launch(void* const* d_in, const int* in_sizes, int n_in,
                              void* d_out, int out_size, void* d_ws, size_t ws_size,
                              hipStream_t stream) {
  const float* obs      = (const float*)d_in[0];
  const float* action   = (const float*)d_in[1];
  const float* W_in     = (const float*)d_in[2];
  const float* b_in     = (const float*)d_in[3];
  const float* W_inproj = (const float*)d_in[4];
  const float* conv_w   = (const float*)d_in[5];
  const float* conv_b   = (const float*)d_in[6];
  const float* dt_bias  = (const float*)d_in[7];
  // d_in[8] = A_log (unused by reference)
  const float* Dvec     = (const float*)d_in[9];
  const float* rms_w    = (const float*)d_in[10];
  const float* W_out    = (const float*)d_in[11];
  const float* W_obs    = (const float*)d_in[12];
  const float* b_obs    = (const float*)d_in[13];
  const float* W_rew    = (const float*)d_in[14];
  const float* b_rew    = (const float*)d_in[15];

  char* w = (char*)d_ws;
  size_t off = 0;
  auto take = [&](size_t bytes) { char* p = w + off; off += (bytes + 255) & ~(size_t)255; return p; };

  float* Xcat = (float*)take((size_t)BATCH * (OBS + ACT) * 4);
  float* X1   = (float*)take((size_t)BATCH * HID * 4);
  float* Zx   = (float*)take((size_t)BATCH * D_PROJ * 4);
  float* Yb   = (float*)take((size_t)BATCH * D_INNER * 4);
  float* Hb   = (float*)take((size_t)BATCH * HID * 4);
  u16t* WtIn   = (u16t*)take((size_t)HID * (OBS + ACT) * 2);
  u16t* WtProj = (u16t*)take((size_t)D_PROJ * HID * 2);
  u16t* WtOut  = (u16t*)take((size_t)HID * D_INNER * 2);
  u16t* WtObs  = (u16t*)take((size_t)OBS * HID * 2);

  float* out_obs = (float*)d_out;
  float* out_rew = out_obs + (size_t)BATCH * OBS;

  // 1. concat inputs
  {
    int total = BATCH * (OBS + ACT);
    concat_kernel<<<(total + 255) / 256, 256, 0, stream>>>(obs, action, Xcat);
  }
  // 2. weight transpose+convert (f32 [K,N] -> bf16 [N,K]); stays hot in 192MB L2
  {
    dim3 blk(32, 8);
    transpose_cvt_kernel<<<dim3(HID / 32, (OBS + ACT) / 32), blk, 0, stream>>>(W_in, WtIn, OBS + ACT, HID);
    transpose_cvt_kernel<<<dim3(D_PROJ / 32, HID / 32), blk, 0, stream>>>(W_inproj, WtProj, HID, D_PROJ);
    transpose_cvt_kernel<<<dim3(HID / 32, D_INNER / 32), blk, 0, stream>>>(W_out, WtOut, D_INNER, HID);
    transpose_cvt_kernel<<<dim3(OBS / 32, HID / 32), blk, 0, stream>>>(W_obs, WtObs, HID, OBS);
  }
  // 3. X1 = Xcat @ W_in + b_in
  gemm_bf16_kernel<<<dim3(HID / BN, BATCH / BM), 256, 0, stream>>>(
      Xcat, WtIn, b_in, X1, BATCH, HID, OBS + ACT);
  // 4. zxbcdt = X1 @ W_inproj
  gemm_bf16_kernel<<<dim3((D_PROJ + BN - 1) / BN, BATCH / BM), 256, 0, stream>>>(
      X1, WtProj, nullptr, Zx, BATCH, D_PROJ, HID);
  // 5. fused elementwise -> y
  mamba_ew_kernel<<<BATCH, 256, 0, stream>>>(Zx, conv_w, conv_b, dt_bias, Dvec, rms_w, Yb);
  // 6. h = y @ W_out
  gemm_bf16_kernel<<<dim3(HID / BN, BATCH / BM), 256, 0, stream>>>(
      Yb, WtOut, nullptr, Hb, BATCH, HID, D_INNER);
  // 7. next_obs = h @ W_obs + b_obs (straight into d_out)
  gemm_bf16_kernel<<<dim3(OBS / BN, BATCH / BM), 256, 0, stream>>>(
      Hb, WtObs, b_obs, out_obs, BATCH, OBS, HID);
  // 8. reward = h @ W_rew + b_rew
  reward_kernel<<<BATCH / 8, 256, 0, stream>>>(Hb, W_rew, b_rew, out_rew);
}
